// CausalBankModel_68126771249607
// MI455X (gfx1250) — compile-verified, matrix operand order checked by reading, not run on hardware
//
#include <hip/hip_runtime.h>
#include <hip/hip_bf16.h>

// ---------------- problem constants ----------------
#define VV 32000
#define EE 256
#define MM 256
#define TT 1024
#define BB 2
#define HH 1024
#define WW 8
#define ROWS (BB * TT)        // 2048
#define LIN_K (MM + EE)       // 512
#define LOC_K (WW * EE)       // 2048

typedef __bf16 bf16_t;
typedef __attribute__((ext_vector_type(16))) __bf16        v16bf;
typedef __attribute__((ext_vector_type(8)))  float         v8f;
typedef __attribute__((ext_vector_type(4)))  int           v4i;
typedef __attribute__((ext_vector_type(4)))  unsigned int  u32x4;
typedef __attribute__((ext_vector_type(8)))  int           i32x8;
typedef __attribute__((ext_vector_type(4)))  int           i32x4;

#if __has_builtin(__builtin_amdgcn_tensor_load_to_lds) && \
    __has_builtin(__builtin_amdgcn_s_wait_tensorcnt)
#define HAVE_TDM 1
#else
#define HAVE_TDM 0
#endif

// ---------------- embed gather: x = emb[chars] -> bf16, and into A_lin[:,256:512] ----
__global__ void k_embed(const int* __restrict__ chars, const float* __restrict__ emb,
                        bf16_t* __restrict__ xbf, bf16_t* __restrict__ Alin) {
    int row = blockIdx.x;           // 0..2047
    int e   = threadIdx.x;          // 0..255
    int c   = chars[row];
    float v = emb[(long)c * EE + e];
    xbf[(long)row * EE + e] = (bf16_t)v;
    Alin[(long)row * LIN_K + MM + e] = (bf16_t)v;
}

// ---------------- weight pack: f32 [K,N] row-major -> bf16 in WMMA B-fragment order ----
// Packed index: (((nt*(K/32) + ks)*32 + lane)*16 + elem)
__global__ void k_pack_w(const float* __restrict__ W, bf16_t* __restrict__ P, int K, int N) {
    long idx   = (long)blockIdx.x * blockDim.x + threadIdx.x;
    long total = (long)(N / 16) * (K / 32) * 32;
    if (idx >= total) return;
    int  lane = (int)(idx & 31);
    long t    = idx >> 5;
    int  ks   = (int)(t % (K / 32));
    int  nt   = (int)(t / (K / 32));
    int  col  = nt * 16 + (lane & 15);
    int  h    = lane >> 4;
    bf16_t* out = P + idx * 16;
    int k0 = ks * 32 + h * 8;
#pragma unroll
    for (int e = 0; e < 8; ++e) out[e]     = (bf16_t)W[(long)(k0 + e) * N + col];
#pragma unroll
    for (int e = 0; e < 8; ++e) out[8 + e] = (bf16_t)W[(long)(k0 + 16 + e) * N + col];
}

#if HAVE_TDM
// TDM: stage one (ntb, ks) B tile = 4 rows x 1KB (row stride ksteps*1KB) into LDS.
// D# per CDNA5 ISA: data_size=3 (8B units), tile 128x4.
__device__ __forceinline__ void tdm_stage(const bf16_t* Bp, long ntb, int ksteps, int ks,
                                          unsigned lds_off) {
    unsigned long long gaddr =
        (unsigned long long)(const void*)(Bp + (ntb * (long)ksteps + ks) * 512);
    unsigned stride0 = (unsigned)ksteps * 128u;          // dim0 stride, 8B units
    u32x4 g0;
    g0[0] = 1u;                                          // count=1 (valid), user mode
    g0[1] = lds_off;                                     // lds_addr [63:32]
    g0[2] = (unsigned)(gaddr & 0xFFFFFFFFull);           // global_addr lo
    g0[3] = (unsigned)((gaddr >> 32) & 0x01FFFFFFull)    // global_addr [56:32]
            | (2u << 30);                                // type=2 ("image")
    i32x8 g1;
    g1[0] = (int)(3u << 16);                             // data_size=3 (8 bytes)
    g1[1] = (int)((stride0 & 0xFFFFu) << 16);            // tensor_dim0 lo16 @[63:48]
    g1[2] = (int)(((stride0 >> 16) & 0xFFFFu)            // tensor_dim0 hi16
                  | (4u << 16));                         // tensor_dim1 lo16 = 4
    g1[3] = (int)(128u << 16);                           // tensor_dim1 hi=0, tile_dim0=128
    g1[4] = (int)4u;                                     // tile_dim1=4, tile_dim2=0
    g1[5] = (int)stride0;                                // tensor_dim0_stride lo32
    g1[6] = 0;                                           // stride0 hi16, stride1 lo16
    g1[7] = 0;
    i32x4 gz = {0, 0, 0, 0};
#if __clang_major__ >= 23
    i32x8 gz8 = {0, 0, 0, 0, 0, 0, 0, 0};
    __builtin_amdgcn_tensor_load_to_lds(g0, g1, gz, gz, gz8, 0);
#else
    __builtin_amdgcn_tensor_load_to_lds(g0, g1, gz, gz, 0);
#endif
}
#endif

// Fallback cooperative staging: 4KB tile, 256 threads x 16B (b128 load + ds store).
__device__ __forceinline__ void stage_b(const bf16_t* __restrict__ Bp, long ntb, int ksteps,
                                        int ks, bf16_t* dst, int tid) {
    int chunk = tid >> 6;             // 0..3 : which 16-wide N tile
    int off   = (tid & 63) * 8;       // element offset within 512
    const v4i* src = (const v4i*)(Bp + ((ntb + chunk) * (long)ksteps + ks) * 512 + off);
    *(v4i*)(dst + chunk * 512 + off) = *src;
}

// ---------------- bf16 WMMA GEMM with TDM/LDS-staged B --------------------------------
// 256 threads = 8 waves; block tile 256(M) x 64(N); wave w covers rows
// [by*256 + w*32, +32) as TWO 16-row A fragments -> 8 WMMAs per 32-K step.
// B tile (4KB per step) staged in LDS, double-buffered via TENSORcnt + barrier.
template <int RELU, int OUT_BF16>
__global__ void k_gemm(const bf16_t* __restrict__ A, const bf16_t* __restrict__ Bp,
                       const float* __restrict__ bias, void* __restrict__ Cout,
                       int N, int K) {
    __shared__ __align__(16) bf16_t shb[2][2048];        // 2 x 4KB
    const int tid  = threadIdx.x;
    const int lane = tid & 31;
    const int wave = tid >> 5;
    const int m0   = blockIdx.y * 256 + wave * 32;
    const int n0   = blockIdx.x * 64;
    const int r    = lane & 15;
    const int h    = lane >> 4;
    const int ksteps = K >> 5;
    const long ntb   = (long)(n0 >> 4);

    v8f zero = {};
    v8f acc0[4] = {zero, zero, zero, zero};
    v8f acc1[4] = {zero, zero, zero, zero};
    const bf16_t* arow0 = A + (long)(m0 + r) * K + h * 8;
    const bf16_t* arow1 = arow0 + 16 * (long)K;

#if HAVE_TDM
    unsigned lds0 = (unsigned)(unsigned long long)(size_t)&shb[0][0];
    unsigned lds1 = (unsigned)(unsigned long long)(size_t)&shb[1][0];
    if (wave == 0) tdm_stage(Bp, ntb, ksteps, 0, lds0);
#else
    stage_b(Bp, ntb, ksteps, 0, shb[0], tid);
#endif

    union F { v16bf v; v4i i[2]; };
    for (int ks = 0; ks < ksteps; ++ks) {
        bf16_t* cur = shb[ks & 1];
#if HAVE_TDM
        if (wave == 0) __builtin_amdgcn_s_wait_tensorcnt(0);   // buf[ks&1] landed
#endif
        __syncthreads();
        if (ks + 1 < ksteps) {
#if HAVE_TDM
            if (wave == 0) tdm_stage(Bp, ntb, ksteps, ks + 1, (ks & 1) ? lds0 : lds1);
#else
            stage_b(Bp, ntb, ksteps, ks + 1, shb[(ks + 1) & 1], tid);
#endif
            __builtin_prefetch(arow0 + (ks + 1) * 32, 0, 0);   // global_prefetch_b8
            __builtin_prefetch(arow1 + (ks + 1) * 32, 0, 0);
        }

        F a0, a1, bf[4];
        const v4i* ap0 = (const v4i*)(arow0 + ks * 32);
        const v4i* ap1 = (const v4i*)(arow1 + ks * 32);
        a0.i[0] = ap0[0];  a0.i[1] = ap0[2];   // K = ks*32 + h*8 (+16) + 0..7
        a1.i[0] = ap1[0];  a1.i[1] = ap1[2];
#pragma unroll
        for (int j = 0; j < 4; ++j) {          // gather all B frags first
            const v4i* bp = (const v4i*)(cur + j * 512 + lane * 16);
            bf[j].i[0] = bp[0];
            bf[j].i[1] = bp[1];
        }
#pragma unroll
        for (int j = 0; j < 4; ++j)
            acc0[j] = __builtin_amdgcn_wmma_f32_16x16x32_bf16(
                false, a0.v, false, bf[j].v, (short)0, acc0[j], false, false);
#pragma unroll
        for (int j = 0; j < 4; ++j)
            acc1[j] = __builtin_amdgcn_wmma_f32_16x16x32_bf16(
                false, a1.v, false, bf[j].v, (short)0, acc1[j], false, false);
    }

#pragma unroll
    for (int j = 0; j < 4; ++j) {
        int   n  = n0 + j * 16 + r;
        float bv = bias ? bias[n] : 0.0f;
#pragma unroll
        for (int i = 0; i < 8; ++i) {
            int   mr0 = m0 + h * 8 + i;
            int   mr1 = mr0 + 16;
            float v0  = acc0[j][i] + bv;
            float v1  = acc1[j][i] + bv;
            if (RELU) { v0 = v0 > 0.0f ? v0 : 0.0f; v1 = v1 > 0.0f ? v1 : 0.0f; }
            if (OUT_BF16) {
                ((bf16_t*)Cout)[(long)mr0 * N + n] = (bf16_t)v0;
                ((bf16_t*)Cout)[(long)mr1 * N + n] = (bf16_t)v1;
            } else {
                ((float*)Cout)[(long)mr0 * N + n] = v0;
                ((float*)Cout)[(long)mr1 * N + n] = v1;
            }
        }
    }
}

// ---------------- decaying state bank: s[t] = d*s[t-1] + drive[t] -> A_lin[:,0:256] ----
__global__ void k_scan(const float* __restrict__ drive, const float* __restrict__ decays,
                       bf16_t* __restrict__ Alin) {
    int idx = blockIdx.x * blockDim.x + threadIdx.x;   // 0..511
    int b   = idx >> 8;
    int m   = idx & 255;
    float d = decays[m];
    float s = 0.0f;
    long base = (long)b * TT;
    for (int t = 0; t < TT; ++t) {
        s = s * d + drive[(base + t) * MM + m];
        Alin[(base + t) * LIN_K + m] = (bf16_t)s;
    }
}

// ---------------- windowed stack: stacked[row, o*E+e] = x[t-o, e] (0 if t-o<0) --------
__global__ void k_stack(const bf16_t* __restrict__ xbf, bf16_t* __restrict__ stacked) {
    int row = blockIdx.x;
    int e   = threadIdx.x;
    int b   = row >> 10;
    int t   = row & (TT - 1);
#pragma unroll
    for (int o = 0; o < WW; ++o) {
        bf16_t v = (bf16_t)0.0f;
        int ts = t - o;
        if (ts >= 0) v = xbf[((long)(b << 10) + ts) * EE + e];
        stacked[(long)row * LOC_K + o * EE + e] = v;
    }
}

// ---------------- per-row logit stats + gate ------------------------------------------
__global__ void k_feats(const float* __restrict__ linL, const float* __restrict__ locL,
                        const float* __restrict__ gW, const float* __restrict__ gB,
                        float* __restrict__ gate) {
    int row = blockIdx.x;
    const float* lr = linL + (long)row * VV;
    const float* cr = locL + (long)row * VV;
    __shared__ float sh[6][256];
    int tid = threadIdx.x;

    float mx1 = -1e30f, mx2 = -1e30f, s1 = 0, s2 = 0, q1 = 0, q2 = 0;
    for (int v = tid; v < VV; v += 256) {
        float a = lr[v], b = cr[v];
        mx1 = fmaxf(mx1, a); mx2 = fmaxf(mx2, b);
        s1 += a; s2 += b; q1 += a * a; q2 += b * b;
    }
    sh[0][tid] = mx1; sh[1][tid] = mx2; sh[2][tid] = s1;
    sh[3][tid] = s2;  sh[4][tid] = q1;  sh[5][tid] = q2;
    __syncthreads();
    for (int off = 128; off > 0; off >>= 1) {
        if (tid < off) {
            sh[0][tid] = fmaxf(sh[0][tid], sh[0][tid + off]);
            sh[1][tid] = fmaxf(sh[1][tid], sh[1][tid + off]);
            sh[2][tid] += sh[2][tid + off];
            sh[3][tid] += sh[3][tid + off];
            sh[4][tid] += sh[4][tid + off];
            sh[5][tid] += sh[5][tid + off];
        }
        __syncthreads();
    }
    float M1 = sh[0][0], M2 = sh[1][0];
    float sum1 = sh[2][0], sum2 = sh[3][0], sq1 = sh[4][0], sq2 = sh[5][0];
    __syncthreads();

    float e1 = 0, w1 = 0, e2 = 0, w2 = 0;
    for (int v = tid; v < VV; v += 256) {
        float a = lr[v] - M1, b = cr[v] - M2;
        float ea = __expf(a), eb = __expf(b);
        e1 += ea; w1 += a * ea; e2 += eb; w2 += b * eb;
    }
    sh[0][tid] = e1; sh[1][tid] = w1; sh[2][tid] = e2; sh[3][tid] = w2;
    __syncthreads();
    for (int off = 128; off > 0; off >>= 1) {
        if (tid < off) {
            sh[0][tid] += sh[0][tid + off];
            sh[1][tid] += sh[1][tid + off];
            sh[2][tid] += sh[2][tid + off];
            sh[3][tid] += sh[3][tid + off];
        }
        __syncthreads();
    }
    if (tid == 0) {
        float S1 = sh[0][0], T1 = sh[1][0], S2 = sh[2][0], T2 = sh[3][0];
        float ent1 = logf(S1) - T1 / S1;
        float ent2 = logf(S2) - T2 / S2;
        float mean1 = sum1 / VV, mean2 = sum2 / VV;
        float var1 = sq1 / VV - mean1 * mean1;
        float var2 = sq2 / VV - mean2 * mean2;
        float z = ent1 * gW[0] + M1 * gW[1] + var1 * gW[2] +
                  ent2 * gW[3] + M2 * gW[4] + var2 * gW[5] + gB[0];
        gate[row] = 1.0f / (1.0f + __expf(-z));
    }
}

// ---------------- final mix: out = g*lin + (1-g)*loc ----------------------------------
__global__ void k_mix(const float* __restrict__ locL, const float* __restrict__ gate,
                      float* __restrict__ out) {
    long i = (long)blockIdx.x * blockDim.x + threadIdx.x;
    long total = (long)ROWS * VV;
    if (i >= total) return;
    int row = (int)(i / VV);
    float g = gate[row];
    out[i] = g * out[i] + (1.0f - g) * locL[i];
}

// =====================================================================================
static inline size_t align256(size_t x) { return (x + 255) & ~(size_t)255; }

extern "C" void kernel_launch(void* const* d_in, const int* in_sizes, int n_in,
                              void* d_out, int out_size, void* d_ws, size_t ws_size,
                              hipStream_t stream) {
    (void)in_sizes; (void)n_in; (void)out_size; (void)ws_size;
    const int*   chars   = (const int*)  d_in[0];
    const float* emb     = (const float*)d_in[1];
    const float* in_proj = (const float*)d_in[2];
    const float* decays  = (const float*)d_in[3];
    const float* lin_W1  = (const float*)d_in[4];
    const float* lin_b1  = (const float*)d_in[5];
    const float* lin_W2  = (const float*)d_in[6];
    const float* lin_b2  = (const float*)d_in[7];
    const float* loc_W1  = (const float*)d_in[8];
    const float* loc_b1  = (const float*)d_in[9];
    const float* loc_W2  = (const float*)d_in[10];
    const float* loc_b2  = (const float*)d_in[11];
    const float* gate_W  = (const float*)d_in[12];
    const float* gate_b  = (const float*)d_in[13];
    float* out = (float*)d_out;

    // workspace carve-up
    char* ws = (char*)d_ws;
    size_t off = 0;
    auto take = [&](size_t bytes) { char* p = ws + off; off += align256(bytes); return p; };
    bf16_t* xbf     = (bf16_t*)take((size_t)ROWS * EE * 2);
    bf16_t* Alin    = (bf16_t*)take((size_t)ROWS * LIN_K * 2);
    bf16_t* stacked = (bf16_t*)take((size_t)ROWS * LOC_K * 2);
    float*  drive   = (float*) take((size_t)ROWS * MM * 4);
    bf16_t* pInP    = (bf16_t*)take((size_t)EE * MM * 2);
    bf16_t* pLinW1  = (bf16_t*)take((size_t)LIN_K * HH * 2);
    bf16_t* pLocW1  = (bf16_t*)take((size_t)LOC_K * HH * 2);
    bf16_t* pLinW2  = (bf16_t*)take((size_t)HH * VV * 2);
    bf16_t* pLocW2  = (bf16_t*)take((size_t)HH * VV * 2);
    bf16_t* hidLin  = (bf16_t*)take((size_t)ROWS * HH * 2);
    bf16_t* hidLoc  = (bf16_t*)take((size_t)ROWS * HH * 2);
    float*  locLog  = (float*) take((size_t)ROWS * VV * 4);
    float*  gate    = (float*) take((size_t)ROWS * 4);

    // 1) embed gather
    k_embed<<<ROWS, 256, 0, stream>>>(chars, emb, xbf, Alin);

    // 2) weight packs (f32 -> bf16 WMMA-B layout)
    auto pack = [&](const float* W, bf16_t* P, int K, int N) {
        long total = (long)(N / 16) * (K / 32) * 32;
        k_pack_w<<<(int)((total + 255) / 256), 256, 0, stream>>>(W, P, K, N);
    };
    pack(in_proj, pInP,   EE,    MM);
    pack(lin_W1,  pLinW1, LIN_K, HH);
    pack(loc_W1,  pLocW1, LOC_K, HH);
    pack(lin_W2,  pLinW2, HH,    VV);
    pack(loc_W2,  pLocW2, HH,    VV);

    // 3) drive = x @ in_proj (f32 out, no bias/relu)
    k_gemm<0, 0><<<dim3(MM / 64, ROWS / 256), 256, 0, stream>>>(
        xbf, pInP, nullptr, drive, MM, EE);

    // 4) decaying recurrence -> A_lin[:,0:256]
    k_scan<<<2, 256, 0, stream>>>(drive, decays, Alin);

    // 5) windowed stack
    k_stack<<<ROWS, 256, 0, stream>>>(xbf, stacked);

    // 6) hidden layers: relu(A@W1 + b1) -> bf16
    k_gemm<1, 1><<<dim3(HH / 64, ROWS / 256), 256, 0, stream>>>(
        Alin, pLinW1, lin_b1, hidLin, HH, LIN_K);
    k_gemm<1, 1><<<dim3(HH / 64, ROWS / 256), 256, 0, stream>>>(
        stacked, pLocW1, loc_b1, hidLoc, HH, LOC_K);

    // 7) logits: hidden @ W2 + b2 (f32). lin -> d_out, loc -> workspace
    k_gemm<0, 0><<<dim3(VV / 64, ROWS / 256), 256, 0, stream>>>(
        hidLin, pLinW2, lin_b2, out, VV, HH);
    k_gemm<0, 0><<<dim3(VV / 64, ROWS / 256), 256, 0, stream>>>(
        hidLoc, pLocW2, loc_b2, locLog, VV, HH);

    // 8) per-row features + gate
    k_feats<<<ROWS, 256, 0, stream>>>(out, locLog, gate_W, gate_b, gate);

    // 9) mix into d_out
    long total = (long)ROWS * VV;
    k_mix<<<(int)((total + 255) / 256), 256, 0, stream>>>(locLog, gate, out);
}
